// MultiHead_25563645346289
// MI455X (gfx1250) — matrix operand order repaired and checked by reading, not compiled, not run on hardware
//
#include <hip/hip_runtime.h>
#include <hip/hip_bf16.h>
#include <cstdint>

// MultiHead attention, CDNA5 (gfx1250), wave32 + v_wmma_f32_16x16x32_f16
// + Tensor Data Mover (tensor_load_to_lds / s_wait_tensorcnt) for K/V tiles.
// B=2, S=2048, H=16, D=64, DM=1024.

#define BB 2
#define SS 2048
#define HH 16
#define DD 64
#define DMODEL 1024

typedef __attribute__((ext_vector_type(16))) _Float16 v16h;
typedef __attribute__((ext_vector_type(8)))  float    v8f;

// ---- TDM availability (probe-verified builtin on ROCm 7.2 & amdgpu-toolchain) ----
#if defined(__has_builtin)
# if __has_builtin(__builtin_amdgcn_tensor_load_to_lds) && \
     __has_builtin(__builtin_amdgcn_s_wait_tensorcnt)
#  define USE_TDM 1
# endif
#endif
#ifndef USE_TDM
# define USE_TDM 0
#endif

__device__ __forceinline__ v8f wmma_f16(v16h a, v16h b, v8f c) {
  // D = A(16x32,f16) * B(32x16,f16) + C(16x16,f32)
  return __builtin_amdgcn_wmma_f32_16x16x32_f16(
      /*neg_a=*/false, a, /*neg_b=*/false, b,
      /*c_mod=*/(short)0, c, /*reuse_a=*/false, /*reuse_b=*/false);
}

// ---- WMMA fragment loaders (contiguous k order: lane half h holds k0+16h .. k0+16h+15) ----

// A fragment 16x32 (MxK) from row-major LDS tile [rows][lda].
__device__ __forceinline__ v16h frag_a(const _Float16* p, int lda, int row0, int k0) {
  int lane = threadIdx.x & 31;
  int half = lane >> 4, m = lane & 15;
  const _Float16* base = p + (row0 + m) * lda + k0 + half * 16;
  v16h a;
#pragma unroll
  for (int e = 0; e < 16; ++e) a[e] = base[e];
  return a;
}

// B fragment 32x16 (KxN) from row-major LDS tile [K][ldb], columns n0..n0+15.
__device__ __forceinline__ v16h frag_b(const _Float16* p, int ldb, int k0, int n0) {
  int lane = threadIdx.x & 31;
  int half = lane >> 4, n = lane & 15;
  const _Float16* base = p + (k0 + half * 16) * ldb + n0 + n;
  v16h b;
#pragma unroll
  for (int e = 0; e < 16; ++e) b[e] = base[e * ldb];
  return b;
}

// B fragment for K^T: source LDS tile is [N][ld]; contiguous reads along a row.
__device__ __forceinline__ v16h frag_bT(const _Float16* p, int ld, int n0, int k0) {
  int lane = threadIdx.x & 31;
  int half = lane >> 4, n = lane & 15;
  const _Float16* base = p + (n0 + n) * ld + k0 + half * 16;
  v16h b;
#pragma unroll
  for (int e = 0; e < 16; ++e) b[e] = base[e];
  return b;
}

#if USE_TDM
typedef __attribute__((ext_vector_type(4))) unsigned int tdm_v4u;
typedef __attribute__((ext_vector_type(8))) int          tdm_v8i;
typedef __attribute__((ext_vector_type(4))) int          tdm_v4i;

// Issue one TDM load of a [tile_h x 64] f16 tile (row-major, row stride 64 elems)
// from global into LDS at byte offset lds_off, with LDS row pitch padded to
// 72 f16: D# pad_interval code 4 (= 2^5 = 32 DWORDs = one 64xf16 row) and
// pad_amount code 3 (= 4 DWORDs = 8 f16).  D# layout per CDNA5 ISA.
__device__ __forceinline__ void tdm_load_tile_f16(unsigned lds_off,
                                                  const _Float16* gsrc,
                                                  unsigned tile_h) {
  unsigned long long ga = (unsigned long long)(size_t)gsrc;
  tdm_v4u g0;
  g0[0] = 1u;                                            // count=1, user desc
  g0[1] = lds_off;                                       // lds_addr (bytes)
  g0[2] = (unsigned)(ga & 0xFFFFFFFFu);                  // global_addr[31:0]
  g0[3] = (unsigned)((ga >> 32) & 0x01FFFFFFu)           // global_addr[56:32]
        | (2u << 30);                                    // type = 2 ("image")
  tdm_v8i g1;
  g1[0] = (int)((1u << 16)        // data_size = 1 -> 2 bytes
              | (1u << 20)        // pad_enable
              | (4u << 22)        // pad_interval code 4 = 32 DWORDs
              | (3u << 25));      // pad_amount  code 3 = 4 DWORDs
  g1[1] = (int)(((unsigned)DD & 0xFFFFu) << 16);         // tensor_dim0 = 64
  g1[2] = (int)(((unsigned)SS & 0xFFFFu) << 16);         // dim0 hi=0 | tensor_dim1 = S
  g1[3] = (int)(((unsigned)DD & 0xFFFFu) << 16);         // dim1 hi=0 | tile_dim0 = 64
  g1[4] = (int)(tile_h & 0xFFFFu);                       // tile_dim1 ; tile_dim2 = 0
  g1[5] = (int)DD;                                       // tensor_dim0_stride = 64
  g1[6] = 0;                                             // stride0 hi | dim1_stride lo
  g1[7] = (int)(((unsigned)(SS * DD)) >> 16);            // dim1_stride hi (unused for 2D)
  tdm_v4i z4 = {0, 0, 0, 0};
#if __clang_major__ >= 23
  tdm_v8i z8 = {0, 0, 0, 0, 0, 0, 0, 0};
  __builtin_amdgcn_tensor_load_to_lds(g0, g1, z4, z4, z8, 0);
#else
  __builtin_amdgcn_tensor_load_to_lds(g0, g1, z4, z4, 0);
#endif
}
#endif  // USE_TDM

// ======================= Kernel 1: QKV projection + RoPE =======================
__global__ __launch_bounds__(128)
void qkv_rope_kernel(const float* __restrict__ q, const float* __restrict__ k,
                     const float* __restrict__ v,
                     const float* __restrict__ Wq, const float* __restrict__ bq,
                     const float* __restrict__ Wk, const float* __restrict__ bk,
                     const float* __restrict__ Wv, const float* __restrict__ bv,
                     _Float16* __restrict__ Qh, _Float16* __restrict__ Kh,
                     _Float16* __restrict__ Vh) {
  const int mode = blockIdx.z;                 // 0=Q, 1=K, 2=V
  const float* X    = mode == 0 ? q  : (mode == 1 ? k  : v);
  const float* W    = mode == 0 ? Wq : (mode == 1 ? Wk : Wv);
  const float* bias = mode == 0 ? bq : (mode == 1 ? bk : bv);
  _Float16*    out  = mode == 0 ? Qh : (mode == 1 ? Kh : Vh);

  __shared__ _Float16 As[64 * 40];
  __shared__ _Float16 Bs[32 * 72];

  const int n0 = blockIdx.x * 64;
  const int m0 = blockIdx.y * 64;
  const int t  = threadIdx.x;
  const int wave = t >> 5;
  const int wm = (wave >> 1) * 32, wn = (wave & 1) * 32;

  v8f c00 = {}, c01 = {}, c10 = {}, c11 = {};

  for (int kk = 0; kk < DMODEL; kk += 32) {
    __syncthreads();
    {   // stage A (f32 -> f16)
      int row = t >> 1, cs = (t & 1) * 16;
      const float* src = X + (size_t)(m0 + row) * DMODEL + kk + cs;
      _Float16* dst = As + row * 40 + cs;
#pragma unroll
      for (int i = 0; i < 16; ++i) dst[i] = (_Float16)src[i];
    }
    {   // stage B (f32 -> f16)
      int row = t >> 2, cs = (t & 3) * 16;
      const float* src = W + (size_t)(kk + row) * DMODEL + n0 + cs;
      _Float16* dst = Bs + row * 72 + cs;
#pragma unroll
      for (int i = 0; i < 16; ++i) dst[i] = (_Float16)src[i];
    }
    __syncthreads();

    v16h a0 = frag_a(As, 40, wm, 0);
    v16h a1 = frag_a(As, 40, wm + 16, 0);
    v16h b0 = frag_b(Bs, 72, 0, wn);
    v16h b1 = frag_b(Bs, 72, 0, wn + 16);
    c00 = wmma_f16(a0, b0, c00);
    c01 = wmma_f16(a0, b1, c01);
    c10 = wmma_f16(a1, b0, c10);
    c11 = wmma_f16(a1, b1, c11);
  }

  // Epilogue: bias + RoPE (pair partner lives in adjacent lane: N = lane%16).
  const int lane = t & 31;
  const int half = lane >> 4, nl = lane & 15;
  v8f* cc[4] = { &c00, &c01, &c10, &c11 };
#pragma unroll
  for (int ij = 0; ij < 4; ++ij) {
    const int mi = ij >> 1, nj = ij & 1;
#pragma unroll
    for (int r = 0; r < 8; ++r) {
      int m = m0 + wm + mi * 16 + half * 8 + r;   // global row (b*S + s)
      int n = n0 + wn + nj * 16 + nl;             // global col (h*D + d)
      float val = (*cc[ij])[r] + bias[n];
      int d = n & (DD - 1);
      int h = n >> 6;
      int s = m & (SS - 1);
      int b = m >> 11;
      float res = val;
      if (mode < 2) {  // RoPE on Q and K
        float other = __shfl_xor(val, 1, 32);     // partner dim d^1
        float inv = __expf(-0.28782313662425575f * (float)d);  // 10000^(-2d/64)
        float ang = (float)s * inv;
        float cs = __cosf(ang), sn = __sinf(ang);
        res = (d & 1) ? (cs * val + sn * other)
                      : (cs * val - sn * other);
      }
      out[(((size_t)b * HH + h) * SS + s) * DD + d] = (_Float16)res;
    }
  }
}

// ======================= Kernel 2: flash attention =======================
// LDS carved from one block so TDM D# lds_addr byte offsets are compile-time.
#define QS_ELEMS (64 * 72)
#define KS_ELEMS (32 * 72)
#define VS_ELEMS (32 * 72)
#define PS_ELEMS (4 * 16 * 40)
#define KS_BYTE_OFF (QS_ELEMS * 2)
#define VS_BYTE_OFF ((QS_ELEMS + KS_ELEMS) * 2)

__global__ __launch_bounds__(128)
void flash_attn_kernel(const _Float16* __restrict__ Qh, const _Float16* __restrict__ Kh,
                       const _Float16* __restrict__ Vh, const float* __restrict__ mask,
                       _Float16* __restrict__ ctx) {
  const int t = threadIdx.x, wave = t >> 5, lane = t & 31;
  const int half = lane >> 4, nl = lane & 15;
  const int q0 = blockIdx.x * 64;
  const int bh = blockIdx.y;
  const int b = bh >> 4, h = bh & 15;

  const _Float16* Qbase = Qh + (size_t)bh * SS * DD;
  const _Float16* Kbase = Kh + (size_t)bh * SS * DD;
  const _Float16* Vbase = Vh + (size_t)bh * SS * DD;
  const float*    mrow  = mask + (size_t)b * SS;

  __shared__ _Float16 smem[QS_ELEMS + KS_ELEMS + VS_ELEMS + PS_ELEMS];
  _Float16* Qs = smem;
  _Float16* Ks = smem + QS_ELEMS;
  _Float16* Vs = smem + QS_ELEMS + KS_ELEMS;
  _Float16* Pw = smem + QS_ELEMS + KS_ELEMS + VS_ELEMS + wave * (16 * 40);

  {   // stage 64x64 Q tile once
    int row = t >> 1, cs = (t & 1) * 32;
    const _Float16* src = Qbase + (size_t)(q0 + row) * DD + cs;
    _Float16* dst = Qs + row * 72 + cs;
#pragma unroll
    for (int i = 0; i < 32; ++i) dst[i] = src[i];
  }
  __syncthreads();
  v16h aq0 = frag_a(Qs, 72, wave * 16, 0);    // d = 0..31
  v16h aq1 = frag_a(Qs, 72, wave * 16, 32);   // d = 32..63

  float m_run[8], l_run[8];
#pragma unroll
  for (int r = 0; r < 8; ++r) { m_run[r] = -1e30f; l_run[r] = 0.f; }
  v8f o0 = {}, o1 = {}, o2 = {}, o3 = {};
  v8f* oo[4] = { &o0, &o1, &o2, &o3 };

  const float inv_scale = 1.0f / (8.0f + 1e-8f);  // 1/(sqrt(D)+1e-8)

  for (int j = 0; j < SS; j += 32) {
    __syncthreads();                               // prior iter's LDS reads done
#if USE_TDM
    if (wave == 0) {
      // TDM: DMA K and V 32x64 f16 tiles into LDS (72-element padded pitch),
      // tracked by TENSORcnt; only the issuing wave waits, barrier releases rest.
      tdm_load_tile_f16(KS_BYTE_OFF, Kbase + (size_t)j * DD, 32);
      tdm_load_tile_f16(VS_BYTE_OFF, Vbase + (size_t)j * DD, 32);
      __builtin_amdgcn_s_wait_tensorcnt(0);
    }
#else
    {   // manual staging fallback
      int row = t >> 2, cs = (t & 3) * 16;
      const _Float16* ksrc = Kbase + (size_t)(j + row) * DD + cs;
      const _Float16* vsrc = Vbase + (size_t)(j + row) * DD + cs;
      _Float16* kdst = Ks + row * 72 + cs;
      _Float16* vdst = Vs + row * 72 + cs;
#pragma unroll
      for (int i = 0; i < 16; ++i) { kdst[i] = ksrc[i]; vdst[i] = vsrc[i]; }
    }
    if (j + 32 < SS) {
      __builtin_prefetch(Kbase + (size_t)(j + 32) * DD, 0, 0);
      __builtin_prefetch(Vbase + (size_t)(j + 32) * DD, 0, 0);
    }
#endif
    __syncthreads();

    // S = Q @ K^T  (16x32 per wave, two 16x16 accs, K-dim = D in two steps)
    v8f s0 = {}, s1 = {};
    s0 = wmma_f16(aq0, frag_bT(Ks, 72, 0,  0),  s0);
    s0 = wmma_f16(aq1, frag_bT(Ks, 72, 0,  32), s0);
    s1 = wmma_f16(aq0, frag_bT(Ks, 72, 16, 0),  s1);
    s1 = wmma_f16(aq1, frag_bT(Ks, 72, 16, 32), s1);

    const float mk0 = mrow[j + nl]      * -1e9f;
    const float mk1 = mrow[j + 16 + nl] * -1e9f;

    // online softmax: each C row lives in one 16-lane half -> shfl_xor tree
    float p0[8], p1[8], corr[8];
#pragma unroll
    for (int r = 0; r < 8; ++r) {
      float x0 = s0[r] * inv_scale + mk0;
      float x1 = s1[r] * inv_scale + mk1;
      float rm = fmaxf(x0, x1);
#pragma unroll
      for (int off = 8; off >= 1; off >>= 1) rm = fmaxf(rm, __shfl_xor(rm, off, 32));
      float mn = fmaxf(m_run[r], rm);
      corr[r] = __expf(m_run[r] - mn);
      m_run[r] = mn;
      p0[r] = __expf(x0 - mn);
      p1[r] = __expf(x1 - mn);
      float rs = p0[r] + p1[r];
#pragma unroll
      for (int off = 8; off >= 1; off >>= 1) rs += __shfl_xor(rs, off, 32);
      l_run[r] = l_run[r] * corr[r] + rs;
    }
#pragma unroll
    for (int tt = 0; tt < 4; ++tt)
#pragma unroll
      for (int r = 0; r < 8; ++r) (*oo[tt])[r] *= corr[r];

    // C-layout -> A-layout via per-wave LDS bounce
#pragma unroll
    for (int r = 0; r < 8; ++r) {
      Pw[(half * 8 + r) * 40 + nl]      = (_Float16)p0[r];
      Pw[(half * 8 + r) * 40 + 16 + nl] = (_Float16)p1[r];
    }
    __syncthreads();

    // O += P @ V  (A = 16x32 probs, B = 32x16 V columns, 4 col tiles)
    v16h ap = frag_a(Pw, 40, 0, 0);
#pragma unroll
    for (int tt = 0; tt < 4; ++tt) {
      v16h bvf = frag_b(Vs, 72, 0, tt * 16);
      *oo[tt] = wmma_f16(ap, bvf, *oo[tt]);
    }
  }

  // finalize: O /= l ; write ctx in merged-head [B][S][DM] layout (f16)
#pragma unroll
  for (int tt = 0; tt < 4; ++tt) {
#pragma unroll
    for (int r = 0; r < 8; ++r) {
      float val = (*oo[tt])[r] / l_run[r];
      int s = q0 + wave * 16 + half * 8 + r;
      int d = tt * 16 + nl;
      ctx[((size_t)b * SS + s) * DMODEL + h * DD + d] = (_Float16)val;
    }
  }
}

// ======================= Kernel 3: output projection =======================
__global__ __launch_bounds__(128)
void out_proj_kernel(const _Float16* __restrict__ ctx, const float* __restrict__ Wo,
                     const float* __restrict__ bo, float* __restrict__ out) {
  __shared__ _Float16 As[64 * 40];
  __shared__ _Float16 Bs[32 * 72];
  const int n0 = blockIdx.x * 64, m0 = blockIdx.y * 64;
  const int t = threadIdx.x, wave = t >> 5, lane = t & 31;
  const int wm = (wave >> 1) * 32, wn = (wave & 1) * 32;

  v8f c00 = {}, c01 = {}, c10 = {}, c11 = {};
  for (int kk = 0; kk < DMODEL; kk += 32) {
    __syncthreads();
    {
      int row = t >> 1, cs = (t & 1) * 16;
      const _Float16* src = ctx + (size_t)(m0 + row) * DMODEL + kk + cs;
      _Float16* dst = As + row * 40 + cs;
#pragma unroll
      for (int i = 0; i < 16; ++i) dst[i] = src[i];
    }
    {
      int row = t >> 2, cs = (t & 3) * 16;
      const float* src = Wo + (size_t)(kk + row) * DMODEL + n0 + cs;
      _Float16* dst = Bs + row * 72 + cs;
#pragma unroll
      for (int i = 0; i < 16; ++i) dst[i] = (_Float16)src[i];
    }
    __syncthreads();
    v16h a0 = frag_a(As, 40, wm, 0);
    v16h a1 = frag_a(As, 40, wm + 16, 0);
    v16h b0 = frag_b(Bs, 72, 0, wn);
    v16h b1 = frag_b(Bs, 72, 0, wn + 16);
    c00 = wmma_f16(a0, b0, c00);
    c01 = wmma_f16(a0, b1, c01);
    c10 = wmma_f16(a1, b0, c10);
    c11 = wmma_f16(a1, b1, c11);
  }

  const int half = lane >> 4, nl = lane & 15;
  v8f* cc[4] = { &c00, &c01, &c10, &c11 };
#pragma unroll
  for (int ij = 0; ij < 4; ++ij) {
    const int mi = ij >> 1, nj = ij & 1;
#pragma unroll
    for (int r = 0; r < 8; ++r) {
      int m = m0 + wm + mi * 16 + half * 8 + r;
      int n = n0 + wn + nj * 16 + nl;
      out[(size_t)m * DMODEL + n] = (*cc[ij])[r] + bo[n];
    }
  }
}

// ======================= host launch =======================
extern "C" void kernel_launch(void* const* d_in, const int* in_sizes, int n_in,
                              void* d_out, int out_size, void* d_ws, size_t ws_size,
                              hipStream_t stream) {
  const float* q    = (const float*)d_in[0];
  const float* k    = (const float*)d_in[1];
  const float* v    = (const float*)d_in[2];
  const float* mask = (const float*)d_in[3];
  const float* Wq   = (const float*)d_in[4];
  const float* bq   = (const float*)d_in[5];
  const float* Wk   = (const float*)d_in[6];
  const float* bk   = (const float*)d_in[7];
  const float* Wv   = (const float*)d_in[8];
  const float* bv   = (const float*)d_in[9];
  const float* Wo   = (const float*)d_in[10];
  const float* bo   = (const float*)d_in[11];

  const size_t TS = (size_t)BB * HH * SS * DD;   // 4M f16 elements per tensor
  _Float16* ws  = (_Float16*)d_ws;
  _Float16* Qh  = ws;
  _Float16* Kh  = ws + TS;
  _Float16* Vh  = ws + 2 * TS;
  _Float16* ct  = ws + 3 * TS;                   // ctx, f16 [B][S][DM]

  dim3 blk(128);
  qkv_rope_kernel<<<dim3(DMODEL / 64, (BB * SS) / 64, 3), blk, 0, stream>>>(
      q, k, v, Wq, bq, Wk, bk, Wv, bv, Qh, Kh, Vh);
  flash_attn_kernel<<<dim3(SS / 64, BB * HH), blk, 0, stream>>>(
      Qh, Kh, Vh, mask, ct);
  out_proj_kernel<<<dim3(DMODEL / 64, (BB * SS) / 64), blk, 0, stream>>>(
      ct, Wo, bo, (float*)d_out);
}
